// TopoBrainComponent_3934190043657
// MI455X (gfx1250) — compile-verified, matrix operand order checked by reading, not run on hardware
//
#include <hip/hip_runtime.h>
#include <hip/hip_bf16.h>
#include <math.h>

typedef __attribute__((ext_vector_type(16))) _Float16 v16h;
typedef __attribute__((ext_vector_type(8)))  _Float16 v8h;
typedef __attribute__((ext_vector_type(8)))  float    v8f;

#define N_GRID   8
#define N_NODES  64
#define N_FEAT   64
#define EMBED    128
#define HIDDEN   256
#define N_ATOMS  6
#define BATCH    4096
#define ROWS     (BATCH * N_NODES)          // 262144 token rows

// ---------------------------------------------------------------------------
// WMMA helpers: f16 16x16x32, f32 accumulate.
// A fragment (row-major [M][K], this lane: row m = lane&15, hi = lane>>4):
//   halves 0..7  -> K = kbase + hi*8 + i
//   halves 8..15 -> K = kbase + 16 + hi*8 + (i-8)
// B fragment identical from Bt stored [N][K] row-major (i.e. weightT).
// ---------------------------------------------------------------------------
__device__ __forceinline__ v16h load_frag(const _Float16* p) {
    v8h lo = *reinterpret_cast<const v8h*>(p);
    v8h hi = *reinterpret_cast<const v8h*>(p + 16);
    v16h r;
#pragma unroll
    for (int i = 0; i < 8; ++i) { r[i] = lo[i]; r[i + 8] = hi[i]; }
    return r;
}

__device__ __forceinline__ v8f wmma_f16(v16h a, v16h b, v8f c) {
    return __builtin_amdgcn_wmma_f32_16x16x32_f16(false, a, false, b,
                                                  (short)0, c, false, false);
}

__device__ __forceinline__ float sigmoidf_(float x) { return 1.f / (1.f + expf(-x)); }

// ---------------------------------------------------------------------------
// Prep 1: normalized masked adjacency (64x64) + zero the h-sum accumulator.
// ---------------------------------------------------------------------------
__global__ void k_prep_adj(const float* __restrict__ adjw, float* __restrict__ adj,
                           float* __restrict__ hsum) {
    int n = threadIdx.x;                     // 64 threads
    if (n == 0) hsum[0] = 0.f;
    int r = n >> 3, c = n & 7;
    float sum = 0.f;
    for (int m = 0; m < N_NODES; ++m) {
        int mr = m >> 3, mc = m & 7;
        int md = abs(r - mr) + abs(c - mc);
        float v = (md == 1) ? sigmoidf_(adjw[n * N_NODES + m]) : 0.f;
        sum += v;
    }
    float inv = 1.f / fmaxf(sum, 1e-6f);
    for (int m = 0; m < N_NODES; ++m) {
        int mr = m >> 3, mc = m & 7;
        int md = abs(r - mr) + abs(c - mc);
        float v = (md == 1) ? sigmoidf_(adjw[n * N_NODES + m]) : 0.f;
        adj[n * N_NODES + m] = v * inv;
    }
}

// Kb[a][e1] = sum_e2 basis[a][e2] * Wk[e2][e1]
__global__ void k_prep_kb(const float* __restrict__ basis, const float* __restrict__ Wk,
                          float* __restrict__ Kb) {
    int idx = blockIdx.x * blockDim.x + threadIdx.x;
    if (idx >= N_ATOMS * EMBED) return;
    int a = idx / EMBED, e1 = idx % EMBED;
    float s = 0.f;
    for (int e2 = 0; e2 < EMBED; ++e2) s += basis[a * EMBED + e2] * Wk[e2 * EMBED + e1];
    Kb[a * EMBED + e1] = s;
}

// Mat[e][a] = (1/(sqrt(E)+1e-8)) * sum_e1 Wq[e][e1] * Kb[a][e1]
__global__ void k_prep_mat(const float* __restrict__ Wq, const float* __restrict__ Kb,
                           float* __restrict__ Mat) {
    int idx = blockIdx.x * blockDim.x + threadIdx.x;
    if (idx >= EMBED * N_ATOMS) return;
    int e = idx / N_ATOMS, a = idx % N_ATOMS;
    float invs = 1.f / (sqrtf((float)EMBED) + 1e-8f);
    float s = 0.f;
    for (int e1 = 0; e1 < EMBED; ++e1) s += Wq[e * EMBED + e1] * Kb[a * EMBED + e1];
    Mat[e * N_ATOMS + a] = s * invs;
}

// Wt[col][f] = sum_{m in nbr(n)} adj[n][m] * W_embed[f][m*128+e]  (col = n*128+e), f16
__global__ void k_prep_wt(const float* __restrict__ adj, const float* __restrict__ We,
                          _Float16* __restrict__ Wt) {
    int idx = blockIdx.x * blockDim.x + threadIdx.x;      // 8192*64
    if (idx >= N_NODES * EMBED * N_FEAT) return;
    int col = idx >> 6, f = idx & 63;
    int n = col >> 7, e = col & 127;
    int r = n >> 3, c = n & 7;
    float s = 0.f;
    if (r > 0)          s += adj[n * N_NODES + (n - 8)] * We[f * 8192 + (n - 8) * 128 + e];
    if (r < N_GRID - 1) s += adj[n * N_NODES + (n + 8)] * We[f * 8192 + (n + 8) * 128 + e];
    if (c > 0)          s += adj[n * N_NODES + (n - 1)] * We[f * 8192 + (n - 1) * 128 + e];
    if (c < N_GRID - 1) s += adj[n * N_NODES + (n + 1)] * We[f * 8192 + (n + 1) * 128 + e];
    Wt[col * N_FEAT + f] = (_Float16)s;
}

// bagg[col] = sum_{m in nbr(n)} adj[n][m] * b_embed[m*128+e]
__global__ void k_prep_bagg(const float* __restrict__ adj, const float* __restrict__ be,
                            float* __restrict__ bagg) {
    int col = blockIdx.x * blockDim.x + threadIdx.x;
    if (col >= N_NODES * EMBED) return;
    int n = col >> 7, e = col & 127;
    int r = n >> 3, c = n & 7;
    float s = 0.f;
    if (r > 0)          s += adj[n * N_NODES + (n - 8)] * be[(n - 8) * 128 + e];
    if (r < N_GRID - 1) s += adj[n * N_NODES + (n + 8)] * be[(n + 8) * 128 + e];
    if (c > 0)          s += adj[n * N_NODES + (n - 1)] * be[(n - 1) * 128 + e];
    if (c < N_GRID - 1) s += adj[n * N_NODES + (n + 1)] * be[(n + 1) * 128 + e];
    bagg[col] = s;
}

__global__ void k_prep_xh(const float* __restrict__ x, _Float16* __restrict__ xh) {
    int i = blockIdx.x * blockDim.x + threadIdx.x;
    if (i < BATCH * N_FEAT) xh[i] = (_Float16)x[i];
}

// W1t[h][e] = W1[e][h] ; W2t[e][h] = W2[h][e]   (f16 transposes, [out][in])
__global__ void k_prep_w1t(const float* __restrict__ W1, _Float16* __restrict__ W1t) {
    int i = blockIdx.x * blockDim.x + threadIdx.x;        // 256*128
    if (i >= HIDDEN * EMBED) return;
    int h = i / EMBED, e = i % EMBED;
    W1t[h * EMBED + e] = (_Float16)W1[e * HIDDEN + h];
}
__global__ void k_prep_w2t(const float* __restrict__ W2, _Float16* __restrict__ W2t) {
    int i = blockIdx.x * blockDim.x + threadIdx.x;        // 128*256
    if (i >= EMBED * HIDDEN) return;
    int e = i / HIDDEN, h = i % HIDDEN;
    W2t[e * HIDDEN + h] = (_Float16)W2[h * EMBED + e];
}

// ---------------------------------------------------------------------------
// GEMM 1: xagg[4096][8192] = xh[4096][64] @ WtT + bagg.  Wave = 16 rows x 128 cols.
// ---------------------------------------------------------------------------
__global__ __launch_bounds__(256) void k_embed_gemm(
    const _Float16* __restrict__ xh, const _Float16* __restrict__ Wt,
    const float* __restrict__ bagg, _Float16* __restrict__ xagg) {
    const int lane = threadIdx.x & 31, wave = threadIdx.x >> 5;
    const int g = blockIdx.x * 8 + wave;          // 16384 waves
    const int strip = g >> 6, cchunk = g & 63;
    const int m = lane & 15, hi = lane >> 4;
    const int r0 = strip * 16, c0 = cchunk * 128;

    v16h a0 = load_frag(xh + (r0 + m) * N_FEAT + 0  + hi * 8);
    v16h a1 = load_frag(xh + (r0 + m) * N_FEAT + 32 + hi * 8);
#pragma unroll
    for (int nt = 0; nt < 8; ++nt) {
        const int col = c0 + nt * 16 + m;         // lane's output column
        const _Float16* bp = Wt + col * N_FEAT + hi * 8;
        v8f acc = {};
        acc = wmma_f16(a0, load_frag(bp), acc);
        acc = wmma_f16(a1, load_frag(bp + 32), acc);
        float bb = bagg[col];
#pragma unroll
        for (int r = 0; r < 8; ++r)
            xagg[(r0 + r + 8 * hi) * 8192 + col] = (_Float16)(acc[r] + bb);
    }
}

// ---------------------------------------------------------------------------
// Fused node MLP: h = relu(xagg@W1+b1) kept in LDS (never hits HBM),
// z = h@W2 (unscaled), plus global sum(h) for the metabolism scalars.
// 4 waves/block, each wave owns a 16-row strip.
// ---------------------------------------------------------------------------
__global__ __launch_bounds__(128) void k_node_mlp(
    const _Float16* __restrict__ xagg, const _Float16* __restrict__ W1t,
    const float* __restrict__ b1, const _Float16* __restrict__ W2t,
    _Float16* __restrict__ z, float* __restrict__ hsum) {
    __shared__ __align__(16) _Float16 ht[4][16][HIDDEN + 8];   // 264-half rows: conflict-free A reloads
    const int lane = threadIdx.x & 31, wave = threadIdx.x >> 5;
    const int m = lane & 15, hi = lane >> 4;
    const int r0 = (blockIdx.x * 4 + wave) * 16;

    // phase 1: h tile [16][256]
    v16h a[4];
#pragma unroll
    for (int kk = 0; kk < 4; ++kk)
        a[kk] = load_frag(xagg + (r0 + m) * EMBED + kk * 32 + hi * 8);
    float lsum = 0.f;
#pragma unroll
    for (int nt = 0; nt < 16; ++nt) {
        const int col = nt * 16 + m;
        v8f acc = {};
#pragma unroll
        for (int kk = 0; kk < 4; ++kk)
            acc = wmma_f16(a[kk], load_frag(W1t + col * EMBED + kk * 32 + hi * 8), acc);
        float bb = b1[col];
#pragma unroll
        for (int r = 0; r < 8; ++r) {
            float v = fmaxf(acc[r] + bb, 0.f);
            lsum += v;
            ht[wave][r + 8 * hi][col] = (_Float16)v;
        }
    }
    asm volatile("s_wait_dscnt 0" ::: "memory");   // own-wave LDS RAW fence

    // phase 2: z = h @ W2   (A fragments re-sourced from LDS)
    v16h a2[8];
#pragma unroll
    for (int kk = 0; kk < 8; ++kk)
        a2[kk] = load_frag(&ht[wave][m][kk * 32 + hi * 8]);
#pragma unroll
    for (int nt = 0; nt < 8; ++nt) {
        const int col = nt * 16 + m;
        v8f acc = {};
#pragma unroll
        for (int kk = 0; kk < 8; ++kk)
            acc = wmma_f16(a2[kk], load_frag(W2t + col * HIDDEN + kk * 32 + hi * 8), acc);
#pragma unroll
        for (int r = 0; r < 8; ++r)
            z[(r0 + r + 8 * hi) * EMBED + col] = (_Float16)acc[r];
    }

    // wave-reduce sum(h) -> one atomic per wave
#pragma unroll
    for (int off = 16; off > 0; off >>= 1) lsum += __shfl_xor(lsum, off, 32);
    if (lane == 0) atomicAdd(hsum, lsum);
}

// ---------------------------------------------------------------------------
// Metabolism + sensitivity scalars collapse to one multiplier c = f1*g.
// ---------------------------------------------------------------------------
__global__ void k_scalar(const float* __restrict__ hsum,
                         const float* __restrict__ mW1, const float* __restrict__ mb1,
                         const float* __restrict__ mW2, const float* __restrict__ mb2,
                         const float* __restrict__ gW1, const float* __restrict__ gb1,
                         const float* __restrict__ gW2, const float* __restrict__ gb2,
                         float* __restrict__ cout) {
    float s = hsum[0] / ((float)ROWS * (float)HIDDEN);
    float acc = mb2[0];
    for (int i = 0; i < 8; ++i) acc += fmaxf(s * mW1[i] + mb1[i], 0.f) * mW2[i];
    float mm = sigmoidf_(acc);
    float state = 0.9f * 0.5f + 0.1f * mm;
    float f1 = 1.f + 0.1f * (state - 0.5f);
    float sg = f1 * s;
    float acc2 = gb2[0];
    for (int i = 0; i < 16; ++i) acc2 += fmaxf(sg * gW1[i] + gb1[i], 0.f) * gW2[i];
    cout[0] = f1 * sigmoidf_(acc2);
}

// ---------------------------------------------------------------------------
// Output: p = c*z + b2 ; attn = p@Mat ; softmax(6) ; out = clip(w@basis,±3).
// 64 rows/block, LDS-staged loads and stores.
// ---------------------------------------------------------------------------
__global__ __launch_bounds__(64) void k_attn_out(
    const _Float16* __restrict__ z, const float* __restrict__ b2,
    const float* __restrict__ Mat, const float* __restrict__ basis,
    const float* __restrict__ cptr, float* __restrict__ out) {
    __shared__ __align__(16) _Float16 zt[64][EMBED + 8];   // 136-half rows
    __shared__ float ot[64][EMBED + 1];                    // 129-float rows
    __shared__ float sb2[EMBED], sMat[EMBED * N_ATOMS], sbasis[N_ATOMS * EMBED];
    const int tid = threadIdx.x;
    const int row0 = blockIdx.x * 64;

    for (int i = tid; i < EMBED; i += 64) sb2[i] = b2[i];
    for (int i = tid; i < EMBED * N_ATOMS; i += 64) { sMat[i] = Mat[i]; sbasis[i] = basis[i]; }
    // cooperative f16 tile load (64 rows x 16 v8h chunks)
    for (int k = tid; k < 64 * 16; k += 64) {
        int row = k >> 4, cc = k & 15;
        *reinterpret_cast<v8h*>(&zt[row][cc * 8]) =
            reinterpret_cast<const v8h*>(z)[(row0 + row) * 16 + cc];
    }
    __syncthreads();

    const float c = cptr[0];
    float att[N_ATOMS];
#pragma unroll
    for (int a = 0; a < N_ATOMS; ++a) att[a] = 0.f;
    for (int e = 0; e < EMBED; ++e) {
        float pv = c * (float)zt[tid][e] + sb2[e];
#pragma unroll
        for (int a = 0; a < N_ATOMS; ++a) att[a] += pv * sMat[e * N_ATOMS + a];
    }
    float mx = att[0];
#pragma unroll
    for (int a = 1; a < N_ATOMS; ++a) mx = fmaxf(mx, att[a]);
    float wsum = 0.f, w[N_ATOMS];
#pragma unroll
    for (int a = 0; a < N_ATOMS; ++a) { w[a] = expf(att[a] - mx); wsum += w[a]; }
    float inv = 1.f / wsum;
#pragma unroll
    for (int a = 0; a < N_ATOMS; ++a) w[a] *= inv;
    for (int e = 0; e < EMBED; ++e) {
        float v = 0.f;
#pragma unroll
        for (int a = 0; a < N_ATOMS; ++a) v += w[a] * sbasis[a * EMBED + e];
        ot[tid][e] = fminf(fmaxf(v, -3.f), 3.f);
    }
    __syncthreads();
    // coalesced store: thread tid writes columns 2*tid and 2*tid+1 of each row
    for (int i = 0; i < 64; ++i)
        out[(row0 + i) * EMBED + tid * 2]     = ot[i][tid * 2],
        out[(row0 + i) * EMBED + tid * 2 + 1] = ot[i][tid * 2 + 1];
}

// ---------------------------------------------------------------------------
extern "C" void kernel_launch(void* const* d_in, const int* in_sizes, int n_in,
                              void* d_out, int out_size, void* d_ws, size_t ws_size,
                              hipStream_t stream) {
    const float* x     = (const float*)d_in[0];
    const float* We    = (const float*)d_in[1];
    const float* be    = (const float*)d_in[2];
    const float* adjw  = (const float*)d_in[3];
    const float* W1    = (const float*)d_in[4];
    const float* b1    = (const float*)d_in[5];
    const float* W2    = (const float*)d_in[6];
    const float* b2    = (const float*)d_in[7];
    const float* basis = (const float*)d_in[8];
    const float* Wq    = (const float*)d_in[9];
    const float* Wk    = (const float*)d_in[10];
    const float* mW1   = (const float*)d_in[11];
    const float* mb1   = (const float*)d_in[12];
    const float* mW2   = (const float*)d_in[13];
    const float* mb2   = (const float*)d_in[14];
    const float* gW1   = (const float*)d_in[15];
    const float* gb1   = (const float*)d_in[16];
    const float* gW2   = (const float*)d_in[17];
    const float* gb2   = (const float*)d_in[18];
    float* out = (float*)d_out;

    char* w = (char*)d_ws;
    auto alloc = [&](size_t bytes) { void* p = (void*)w; w += (bytes + 255) & ~(size_t)255; return p; };
    float*    adj   = (float*)   alloc(N_NODES * N_NODES * 4);
    float*    hsum  = (float*)   alloc(256);
    float*    cbuf  = (float*)   alloc(256);
    float*    Kb    = (float*)   alloc(N_ATOMS * EMBED * 4);
    float*    Mat   = (float*)   alloc(EMBED * N_ATOMS * 4);
    float*    bagg  = (float*)   alloc(N_NODES * EMBED * 4);
    _Float16* xh    = (_Float16*)alloc((size_t)BATCH * N_FEAT * 2);
    _Float16* Wt    = (_Float16*)alloc((size_t)N_NODES * EMBED * N_FEAT * 2);
    _Float16* W1t   = (_Float16*)alloc((size_t)HIDDEN * EMBED * 2);
    _Float16* W2t   = (_Float16*)alloc((size_t)EMBED * HIDDEN * 2);
    _Float16* xagg  = (_Float16*)alloc((size_t)ROWS * EMBED * 2);
    _Float16* zbuf  = (_Float16*)alloc((size_t)ROWS * EMBED * 2);

    k_prep_adj <<<1, 64, 0, stream>>>(adjw, adj, hsum);
    k_prep_kb  <<<3, 256, 0, stream>>>(basis, Wk, Kb);
    k_prep_mat <<<3, 256, 0, stream>>>(Wq, Kb, Mat);
    k_prep_wt  <<<(N_NODES * EMBED * N_FEAT + 255) / 256, 256, 0, stream>>>(adj, We, Wt);
    k_prep_bagg<<<(N_NODES * EMBED + 255) / 256, 256, 0, stream>>>(adj, be, bagg);
    k_prep_xh  <<<(BATCH * N_FEAT + 255) / 256, 256, 0, stream>>>(x, xh);
    k_prep_w1t <<<(HIDDEN * EMBED + 255) / 256, 256, 0, stream>>>(W1, W1t);
    k_prep_w2t <<<(EMBED * HIDDEN + 255) / 256, 256, 0, stream>>>(W2, W2t);

    k_embed_gemm<<<2048, 256, 0, stream>>>(xh, Wt, bagg, xagg);
    k_node_mlp  <<<4096, 128, 0, stream>>>(xagg, W1t, b1, W2t, zbuf, hsum);
    k_scalar    <<<1, 1, 0, stream>>>(hsum, mW1, mb1, mW2, mb2, gW1, gb1, gW2, gb2, cbuf);
    k_attn_out  <<<ROWS / 64, 64, 0, stream>>>(zbuf, b2, Mat, basis, cbuf, out);
    (void)in_sizes; (void)n_in; (void)out_size; (void)ws_size;
}
